// SoftDistanceLoss_44263932953086
// MI455X (gfx1250) — matrix operand order repaired and checked by reading, not compiled, not run on hardware
//
#include <hip/hip_runtime.h>
#include <hip/hip_bf16.h>

typedef __attribute__((ext_vector_type(16))) __bf16 v16bf;
typedef __attribute__((ext_vector_type(8)))  float  v8f;
typedef int v4int __attribute__((vector_size(16)));
typedef __attribute__((address_space(1))) v4int gv4;   // global v4i
typedef __attribute__((address_space(3))) v4int lv4;   // LDS v4i

#define B_  8
#define C_  512
#define T_  1500
#define K_  4096
#define N_  (B_ * T_)        // 12000
#define NT_ (N_ / 16)        // 750 row tiles
#define KTILES_ (K_ / 16)    // 256 column tiles
#define CSTEPS_ (C_ / 32)    // 16 wmma k-steps

#define WAVES_   3           // waves (row tiles) per block
#define THREADS_ (WAVES_ * 32)
#define BLOCKS_  (NT_ / WAVES_)   // 250

#define TILE_F4_ 1024        // one A tile = 16KB = 1024 float4 chunks
#define CHUNKS_  ((TILE_F4_ + THREADS_ - 1) / THREADS_)  // 11

// ---- workspace layout (bytes, all 32B aligned) ----
#define OFF_BFRAG 0                                  // feats fragments (bf16)
#define SZ_BFRAG  ((size_t)N_ * C_ * 2)              // 12,288,000
#define OFF_AFRAG (OFF_BFRAG + SZ_BFRAG)             // codebook fragments (bf16)
#define SZ_AFRAG  ((size_t)K_ * C_ * 2)              // 4,194,304
#define OFF_CNORM (OFF_AFRAG + SZ_AFRAG)
#define OFF_FNORM (OFF_CNORM + (size_t)K_ * 4)
#define OFF_ROWED (OFF_FNORM + (size_t)N_ * 4)
#define OFF_ROWMA (OFF_ROWED + (size_t)N_ * 4)
#define OFF_ROWEN (OFF_ROWMA + (size_t)N_ * 4)

// ---------------- prep: feats -> B-fragment order (bf16) ----------------
__global__ __launch_bounds__(32) void prep_feats(const float* __restrict__ sf,
                                                 __bf16* __restrict__ bfr) {
  int id   = blockIdx.x;            // nTile*16 + cStep
  int nT   = id >> 4, cs = id & 15;
  int lane = threadIdx.x;
  int n = nT * 16 + (lane & 15);
  int b = n / T_, t = n % T_;
  int cbase = cs * 32 + ((lane >= 16) ? 16 : 0);
  v16bf v;
#pragma unroll
  for (int h = 0; h < 16; ++h)
    v[h] = (__bf16)sf[((size_t)b * C_ + cbase + h) * T_ + t];
  ((v16bf*)bfr)[(size_t)id * 32 + lane] = v;
}

// ---------------- prep: codebook -> A-fragment order (bf16) ----------------
__global__ __launch_bounds__(32) void prep_cb(const float* __restrict__ cb,
                                              __bf16* __restrict__ afr) {
  int id   = blockIdx.x;            // kTile*16 + cStep
  int kT   = id >> 4, cs = id & 15;
  int lane = threadIdx.x;
  int k = kT * 16 + (lane & 15);
  int off = (lane >= 16) ? 16 : 0;
  v16bf v;
#pragma unroll
  for (int h = 0; h < 16; ++h) {
    int Kc = h + off;
    v[h] = (__bf16)cb[(size_t)k * C_ + cs * 32 + Kc];
  }
  ((v16bf*)afr)[(size_t)id * 32 + lane] = v;
}

// ---------------- norms ----------------
__global__ __launch_bounds__(32) void fnorm_k(const float* __restrict__ sf,
                                              float* __restrict__ fn) {
  int n = blockIdx.x, lane = threadIdx.x;
  int b = n / T_, t = n % T_;
  float s = 0.f;
  for (int c = lane; c < C_; c += 32) {
    float v = sf[((size_t)b * C_ + c) * T_ + t];
    s += v * v;
  }
  for (int o = 16; o > 0; o >>= 1) s += __shfl_down(s, o, 32);
  if (lane == 0) fn[n] = s;
}

__global__ __launch_bounds__(32) void cnorm_k(const float* __restrict__ cb,
                                              float* __restrict__ cn) {
  int k = blockIdx.x, lane = threadIdx.x;
  float s = 0.f;
  for (int c = lane; c < C_; c += 32) {
    float v = cb[(size_t)k * C_ + c];
    s += v * v;
  }
  for (int o = 16; o > 0; o >>= 1) s += __shfl_down(s, o, 32);
  if (lane == 0) cn[k] = s;
}

// ---------------- fused GEMM + flash-softmax + reductions ----------------
// Block = 3 waves, each wave owns a 16-row tile of N. The codebook A tile for
// the current 16 columns lives in double-buffered LDS shared by the 3 waves;
// the NEXT tile is brought in by async global->LDS copies (no VGPR staging,
// ASYNCcnt-tracked) issued a full tile ahead of use.
__global__ __launch_bounds__(THREADS_) void sdl_main(
    const float* __restrict__ dm, const int* __restrict__ codes,
    const __bf16* __restrict__ bfr, const __bf16* __restrict__ afr,
    const float* __restrict__ fnorm, const float* __restrict__ cnorm,
    float* __restrict__ rowEd, float* __restrict__ rowMa,
    float* __restrict__ rowEn) {
  __shared__ float4 ldsA[2][TILE_F4_];        // 2 x 16KB

  int tid  = threadIdx.x;
  int lane = tid & 31;
  int wave = tid >> 5;
  int nTile = blockIdx.x * WAVES_ + wave;     // 250 * 3 = 750
  int col = lane & 15;
  int hi  = lane >> 4;                        // 0: k 0..7 of tile, 1: k 8..15
  int n   = nTile * 16 + col;
  int code = codes[n];
  float fn = fnorm[n];

  const v16bf*  bp   = (const v16bf*)bfr;
  const float4* asrc = (const float4*)afr;

  v16bf bB[CSTEPS_];                          // feats fragments resident
#pragma unroll
  for (int s = 0; s < CSTEPS_; ++s)
    bB[s] = bp[(size_t)(nTile * CSTEPS_ + s) * 32 + lane];

  // async-stage tile 0
#pragma unroll
  for (int j = 0; j < CHUNKS_; ++j) {
    int idx = tid + j * THREADS_;
    if (idx < TILE_F4_)
      __builtin_amdgcn_global_load_async_to_lds_b128(
          (gv4*)(asrc + idx), (lv4*)&ldsA[0][idx], 0, 0);
  }
  asm volatile("s_wait_asynccnt 0" ::: "memory");
  __syncthreads();

  float m = -__builtin_inff(), S = 0.f, W = 0.f, Ed = 0.f;
  float bestL = -__builtin_inff();
  int bestK = -1;

  for (int kt = 0; kt < KTILES_; ++kt) {
    int cur = kt & 1;

    // issue async copies of the next A tile into the back buffer
    if (kt + 1 < KTILES_) {
      const float4* src = asrc + (size_t)(kt + 1) * TILE_F4_;
#pragma unroll
      for (int j = 0; j < CHUNKS_; ++j) {
        int idx = tid + j * THREADS_;
        if (idx < TILE_F4_)
          __builtin_amdgcn_global_load_async_to_lds_b128(
              (gv4*)(src + idx), (lv4*)&ldsA[cur ^ 1][idx], 0, 0);
      }
    }

    // 16 WMMAs from LDS, two interleaved accumulator chains
    const v16bf* aT = (const v16bf*)&ldsA[cur][0];
    v8f acc0 = {0.f, 0.f, 0.f, 0.f, 0.f, 0.f, 0.f, 0.f};
    v8f acc1 = {0.f, 0.f, 0.f, 0.f, 0.f, 0.f, 0.f, 0.f};
#pragma unroll
    for (int s = 0; s < CSTEPS_; s += 2) {
      v16bf a0 = aT[s * 32 + lane];
      v16bf a1 = aT[(s + 1) * 32 + lane];
      acc0 = __builtin_amdgcn_wmma_f32_16x16x32_bf16(
          false, a0, false, bB[s], (short)0, acc0, false, false);
      acc1 = __builtin_amdgcn_wmma_f32_16x16x32_bf16(
          false, a1, false, bB[s + 1], (short)0, acc1, false, false);
    }
    v8f acc = acc0 + acc1;

    // per-lane flash-softmax update over this tile's 8 k values
    int kbase = kt * 16 + hi * 8;
    const float4* cp = (const float4*)(cnorm + kbase);
    float4 c0 = cp[0], c1 = cp[1];
    const float4* tp = (const float4*)(dm + (size_t)code * K_ + kbase);
    float4 t0 = tp[0], t1 = tp[1];
    float cns[8] = {c0.x, c0.y, c0.z, c0.w, c1.x, c1.y, c1.z, c1.w};
    float tds[8] = {t0.x, t0.y, t0.z, t0.w, t1.x, t1.y, t1.z, t1.w};
#pragma unroll
    for (int r = 0; r < 8; ++r) {
      float sq = fn + cns[r] - 2.f * acc[r];
      float l = -__builtin_sqrtf(fmaxf(sq, 0.f));   // logit = -distance
      if (l > bestL) { bestL = l; bestK = kbase + r; }
      if (l > m) {                                   // rescale (rare)
        float sc = __expf(m - l);
        S *= sc; W *= sc; Ed *= sc; m = l;
      }
      float e = __expf(l - m);
      S += e; W += e * l; Ed += e * tds[r];
    }

    // next tile must have landed; publish to all waves
    asm volatile("s_wait_asynccnt 0" ::: "memory");
    __syncthreads();
  }

  // merge with partner lane (other 8 k's of each tile)
  float m2 = __shfl_xor(m, 16, 32);
  float S2 = __shfl_xor(S, 16, 32);
  float W2 = __shfl_xor(W, 16, 32);
  float E2 = __shfl_xor(Ed, 16, 32);
  float bL2 = __shfl_xor(bestL, 16, 32);
  int   bK2 = __shfl_xor(bestK, 16, 32);
  float M = fmaxf(m, m2);
  float s1 = __expf(m - M), s2 = __expf(m2 - M);
  S = S * s1 + S2 * s2;
  W = W * s1 + W2 * s2;
  Ed = Ed * s1 + E2 * s2;
  if (bL2 > bestL || (bL2 == bestL && bK2 < bestK)) { bestL = bL2; bestK = bK2; }

  if (lane < 16) {
    rowEd[n] = Ed / S;
    rowMa[n] = (bestK == code) ? 1.f : 0.f;
    rowEn[n] = M + __logf(S) - W / S;   // -(sum p log p)
  }
}

// ---------------- deterministic final reduction ----------------
__global__ __launch_bounds__(256) void finalize_k(
    const float* __restrict__ rowEd, const float* __restrict__ rowMa,
    const float* __restrict__ rowEn, float* __restrict__ out) {
  __shared__ float sE[256], sM[256], sH[256];
  int t = threadIdx.x;
  float e = 0.f, ma = 0.f, h = 0.f;
  for (int i = t; i < N_; i += 256) {
    e += rowEd[i]; ma += rowMa[i]; h += rowEn[i];
  }
  sE[t] = e; sM[t] = ma; sH[t] = h;
  __syncthreads();
  for (int s = 128; s > 0; s >>= 1) {
    if (t < s) { sE[t] += sE[t + s]; sM[t] += sM[t + s]; sH[t] += sH[t + s]; }
    __syncthreads();
  }
  if (t == 0) {
    out[0] = sE[0] / (float)N_;
    out[1] = sM[0] / (float)N_;
    out[2] = sH[0] / (float)N_;
  }
}

extern "C" void kernel_launch(void* const* d_in, const int* in_sizes, int n_in,
                              void* d_out, int out_size, void* d_ws, size_t ws_size,
                              hipStream_t stream) {
  const float* sf    = (const float*)d_in[0];   // (B, C, T) f32
  const int*   codes = (const int*)d_in[1];     // (1, B, T) int
  const float* cb    = (const float*)d_in[2];   // (K, C) f32
  const float* dm    = (const float*)d_in[3];   // (K, K) f32

  char* ws = (char*)d_ws;
  __bf16* bfr  = (__bf16*)(ws + OFF_BFRAG);
  __bf16* afr  = (__bf16*)(ws + OFF_AFRAG);
  float* cn    = (float*)(ws + OFF_CNORM);
  float* fn    = (float*)(ws + OFF_FNORM);
  float* rowEd = (float*)(ws + OFF_ROWED);
  float* rowMa = (float*)(ws + OFF_ROWMA);
  float* rowEn = (float*)(ws + OFF_ROWEN);

  prep_feats<<<NT_ * CSTEPS_, 32, 0, stream>>>(sf, bfr);   // 12000 blocks
  prep_cb<<<KTILES_ * CSTEPS_, 32, 0, stream>>>(cb, afr);  // 4096 blocks
  fnorm_k<<<N_, 32, 0, stream>>>(sf, fn);
  cnorm_k<<<K_, 32, 0, stream>>>(cb, cn);
  sdl_main<<<BLOCKS_, THREADS_, 0, stream>>>(dm, codes, bfr, afr, fn, cn,
                                             rowEd, rowMa, rowEn);
  finalize_k<<<1, 256, 0, stream>>>(rowEd, rowMa, rowEn, (float*)d_out);
}
